// MultiHeadAttention_54322746360278
// MI455X (gfx1250) — matrix operand order repaired and checked by reading, not compiled, not run on hardware
//
#include <hip/hip_runtime.h>
#include <cstdint>

#define D_MODEL  1024
#define N_HEADS  16
#define HEAD_DIM 64
#define BATCH    2
#define SEQ      2048

typedef __bf16 bf16_t;
typedef __attribute__((ext_vector_type(16))) __bf16 v16bf;
typedef __attribute__((ext_vector_type(8)))  __bf16 v8bf;
typedef __attribute__((ext_vector_type(8)))  float  v8f;

#define WMMA_BF16(a_, b_, c_) \
  __builtin_amdgcn_wmma_f32_16x16x32_bf16(false, (a_), false, (b_), (short)0, (c_), false, false)

// ---- Fragment loaders (CDNA5 wave32 WMMA VGPR layouts, CDNA5 ISA) ----

// A-matrix 16x32 bf16: lane l<16 -> row M=l, halves 0..7 K=0..7, halves 8..15 K=16..23;
// lanes 16..31 same rows with K offset +8. Two 16-byte loads per lane.
__device__ __forceinline__ v16bf load_a_frag(const bf16_t* A, int lda, int m0, int k0, int lane) {
  const int l16 = lane & 15, half = lane >> 4;
  const bf16_t* p = A + (size_t)(m0 + l16) * lda + k0 + half * 8;
  v8bf lo = *(const v8bf*)(p);
  v8bf hi = *(const v8bf*)(p + 16);
  v16bf a;
#pragma unroll
  for (int i = 0; i < 8; ++i) { a[i] = lo[i]; a[i + 8] = hi[i]; }
  return a;
}

// B-matrix 32x16 bf16 (B[k][n] supplied as Bsrc[n][k], i.e. row n contiguous along k):
// lane l<16 -> col N=l, halves 0..7 K=0..7, halves 8..15 K=16..23; lanes 16..31 K offset +8. Two 16-byte loads per lane.
__device__ __forceinline__ v16bf load_b_frag(const bf16_t* B, int ldb, int n0, int k0, int lane) {
  const int l16 = lane & 15, half = lane >> 4;
  const bf16_t* p = B + (size_t)(n0 + l16) * ldb + k0 + half * 8;
  v8bf lo = *(const v8bf*)(p);
  v8bf hi = *(const v8bf*)(p + 16);
  v16bf b;
#pragma unroll
  for (int i = 0; i < 8; ++i) { b[i] = lo[i]; b[i + 8] = hi[i]; }
  return b;
}

// C-fragment rows are spread across the 16 lanes of each half-wave; reduce across N.
__device__ __forceinline__ float rowmax16(float x) {
  x = fmaxf(x, __shfl_xor(x, 1, 32));
  x = fmaxf(x, __shfl_xor(x, 2, 32));
  x = fmaxf(x, __shfl_xor(x, 4, 32));
  x = fmaxf(x, __shfl_xor(x, 8, 32));
  return x;
}
__device__ __forceinline__ float rowsum16(float x) {
  x += __shfl_xor(x, 1, 32);
  x += __shfl_xor(x, 2, 32);
  x += __shfl_xor(x, 4, 32);
  x += __shfl_xor(x, 8, 32);
  return x;
}

// ---- Kernel 1: f32 -> bf16 conversion ----
__global__ void f32_to_bf16_kernel(const float* __restrict__ src, bf16_t* __restrict__ dst, int n) {
  int i = blockIdx.x * blockDim.x + threadIdx.x;
  if (i < n) dst[i] = (bf16_t)src[i];
}

// ---- Kernel 2: out = A @ B^T + bias via v_wmma_f32_16x16x32_bf16 ----
// Each wave owns a 32x64 output tile: 2 A-frags + 4 B-frags -> 8 WMMAs per K-step
// (24 B/lane of fragment traffic per WMMA instead of 64 B/lane for a 16x16 tile).
// mode 1: write bf16 to [b][h][s][d]   (per-head layout for Q/K)
// mode 2: write bf16 to [b][h][d][s]   (transposed per-head layout for V)
// mode 3: write f32  to [m][n]         (final output)
__global__ void __launch_bounds__(128)
gemm_bf16_wmma_kernel(const bf16_t* __restrict__ A, const bf16_t* __restrict__ B,
                      const float* __restrict__ bias, bf16_t* __restrict__ outb,
                      float* __restrict__ outf, int M, int N, int K, int mode) {
  const int lane = threadIdx.x & 31;
  const int wave = blockIdx.x * 4 + (threadIdx.x >> 5);
  const int tilesN = N >> 6;                      // 64-wide N stripes
  const int mt = wave / tilesN, nt = wave - mt * tilesN;
  const int m0 = mt * 32, n0 = nt * 64;
  if (m0 >= M) return;   // wave-uniform

  v8f acc[2][4];
#pragma unroll
  for (int i = 0; i < 2; ++i)
#pragma unroll
    for (int j = 0; j < 4; ++j) acc[i][j] = (v8f){};

  for (int k0 = 0; k0 < K; k0 += 32) {
    v16bf a0 = load_a_frag(A, K, m0,      k0, lane);
    v16bf a1 = load_a_frag(A, K, m0 + 16, k0, lane);
    v16bf b0 = load_b_frag(B, K, n0,      k0, lane);
    v16bf b1 = load_b_frag(B, K, n0 + 16, k0, lane);
    v16bf b2 = load_b_frag(B, K, n0 + 32, k0, lane);
    v16bf b3 = load_b_frag(B, K, n0 + 48, k0, lane);
    acc[0][0] = WMMA_BF16(a0, b0, acc[0][0]);
    acc[0][1] = WMMA_BF16(a0, b1, acc[0][1]);
    acc[0][2] = WMMA_BF16(a0, b2, acc[0][2]);
    acc[0][3] = WMMA_BF16(a0, b3, acc[0][3]);
    acc[1][0] = WMMA_BF16(a1, b0, acc[1][0]);
    acc[1][1] = WMMA_BF16(a1, b1, acc[1][1]);
    acc[1][2] = WMMA_BF16(a1, b2, acc[1][2]);
    acc[1][3] = WMMA_BF16(a1, b3, acc[1][3]);
  }

  const int l16 = lane & 15, half = lane >> 4;
#pragma unroll
  for (int j = 0; j < 4; ++j) {
    const int ncol = n0 + j * 16 + l16;
    const float bv = bias[ncol];
#pragma unroll
    for (int i = 0; i < 2; ++i) {
#pragma unroll
      for (int r = 0; r < 8; ++r) {
        const int mrow = m0 + i * 16 + half * 8 + r;
        const float val = acc[i][j][r] + bv;
        if (mode == 3) {
          outf[(size_t)mrow * N + ncol] = val;
        } else {
          const int bidx = mrow / SEQ, srow = mrow % SEQ;
          const int hh = ncol >> 6, dd = ncol & 63;
          size_t idx;
          if (mode == 1) idx = (((size_t)bidx * N_HEADS + hh) * SEQ + srow) * HEAD_DIM + dd;
          else           idx = (((size_t)bidx * N_HEADS + hh) * HEAD_DIM + dd) * SEQ + srow;
          outb[idx] = (bf16_t)val;
        }
      }
    }
  }
}

// ---- Kernel 3: fused flash attention (one 16-query tile per wave) ----
__global__ void __launch_bounds__(128)
mha_flash_attn_kernel(const bf16_t* __restrict__ Qh, const bf16_t* __restrict__ Kh,
                      const bf16_t* __restrict__ Vt, const int* __restrict__ mask,
                      bf16_t* __restrict__ joint) {
  __shared__ bf16_t plds[4][16 * 32];   // per-wave P-tile bounce buffer (C-layout -> A-layout)
  const int lane = threadIdx.x & 31;
  const int wid  = threadIdx.x >> 5;
  const int wave = blockIdx.x * 4 + wid;
  const int qt = wave & (SEQ / 16 - 1);
  const int bh = wave >> 7;             // 0..31
  const int b  = bh >> 4;
  const int h  = bh & 15;
  const int q0 = qt * 16;
  const int l16 = lane & 15, half = lane >> 4;

  const bf16_t* Qp = Qh + (size_t)bh * SEQ * HEAD_DIM;
  const bf16_t* Kp = Kh + (size_t)bh * SEQ * HEAD_DIM;
  const bf16_t* Vp = Vt + (size_t)bh * HEAD_DIM * SEQ;
  const int*    mp = mask + (size_t)b * SEQ * SEQ;

  // Q tile A-fragments for full head dim (K = 0..31, 32..63)
  v16bf qa0 = load_a_frag(Qp, HEAD_DIM, q0, 0, lane);
  v16bf qa1 = load_a_frag(Qp, HEAD_DIM, q0, 32, lane);

  v8f o0 = {}, o1 = {}, o2 = {}, o3 = {};   // 16x64 f32 accumulator
  float m[8], l[8];
#pragma unroll
  for (int r = 0; r < 8; ++r) { m[r] = -3.0e38f; l[r] = 0.f; }

  bf16_t* pl = &plds[wid][0];

  for (int kb = 0; kb < SEQ; kb += 32) {
    // ---- scores: S = Q @ K^T for 32 keys (two 16x16 C tiles) ----
    v8f s0 = {}, s1 = {};
    {
      v16bf kb00 = load_b_frag(Kp, HEAD_DIM, kb,      0,  lane);
      v16bf kb01 = load_b_frag(Kp, HEAD_DIM, kb,      32, lane);
      v16bf kb10 = load_b_frag(Kp, HEAD_DIM, kb + 16, 0,  lane);
      v16bf kb11 = load_b_frag(Kp, HEAD_DIM, kb + 16, 32, lane);
      s0 = WMMA_BF16(qa0, kb00, s0);
      s0 = WMMA_BF16(qa1, kb01, s0);
      s1 = WMMA_BF16(qa0, kb10, s1);
      s1 = WMMA_BF16(qa1, kb11, s1);
    }

    // ---- scale, mask, online softmax ----
    float alph[8];
#pragma unroll
    for (int r = 0; r < 8; ++r) {
      const int q = q0 + half * 8 + r;
      float v0 = s0[r] * 0.125f;   // 1/sqrt(64)
      float v1 = s1[r] * 0.125f;
      if (mp[(size_t)q * SEQ + kb + l16]      == 0) v0 = -1.0e9f;
      if (mp[(size_t)q * SEQ + kb + 16 + l16] == 0) v1 = -1.0e9f;
      const float mx = rowmax16(fmaxf(v0, v1));
      const float mn = fmaxf(m[r], mx);
      const float al = __expf(m[r] - mn);
      const float e0 = __expf(v0 - mn);
      const float e1 = __expf(v1 - mn);
      l[r] = l[r] * al + rowsum16(e0 + e1);
      m[r] = mn;
      alph[r] = al;
      // P tile to LDS in [row][key] order (row = 8*half + r, key = l16 / l16+16)
      pl[(half * 8 + r) * 32 + l16]      = (bf16_t)e0;
      pl[(half * 8 + r) * 32 + 16 + l16] = (bf16_t)e1;
    }
#pragma unroll
    for (int r = 0; r < 8; ++r) {
      o0[r] *= alph[r]; o1[r] *= alph[r]; o2[r] *= alph[r]; o3[r] *= alph[r];
    }

    // ---- O += P @ V (P re-read from LDS in A-layout; LDS in-order within wave) ----
    v16bf pa  = load_a_frag(pl, 32, 0, 0, lane);
    v16bf vb0 = load_b_frag(Vp, SEQ,  0, kb, lane);
    v16bf vb1 = load_b_frag(Vp, SEQ, 16, kb, lane);
    v16bf vb2 = load_b_frag(Vp, SEQ, 32, kb, lane);
    v16bf vb3 = load_b_frag(Vp, SEQ, 48, kb, lane);
    o0 = WMMA_BF16(pa, vb0, o0);
    o1 = WMMA_BF16(pa, vb1, o1);
    o2 = WMMA_BF16(pa, vb2, o2);
    o3 = WMMA_BF16(pa, vb3, o3);
  }

  // ---- normalize and write joint [b][s][d_model] as bf16 ----
#pragma unroll
  for (int r = 0; r < 8; ++r) {
    const int srow = q0 + half * 8 + r;
    const float inv = 1.0f / l[r];
    const size_t base = ((size_t)b * SEQ + srow) * D_MODEL + h * HEAD_DIM + l16;
    joint[base +  0] = (bf16_t)(o0[r] * inv);
    joint[base + 16] = (bf16_t)(o1[r] * inv);
    joint[base + 32] = (bf16_t)(o2[r] * inv);
    joint[base + 48] = (bf16_t)(o3[r] * inv);
  }
}

extern "C" void kernel_launch(void* const* d_in, const int* in_sizes, int n_in,
                              void* d_out, int out_size, void* d_ws, size_t ws_size,
                              hipStream_t stream) {
  (void)in_sizes; (void)n_in; (void)out_size; (void)ws_size;
  const float* q  = (const float*)d_in[0];
  const float* k  = (const float*)d_in[1];
  const float* v  = (const float*)d_in[2];
  const int*  msk = (const int*)  d_in[3];
  const float* Wq = (const float*)d_in[4];
  const float* bq = (const float*)d_in[5];
  const float* Wk = (const float*)d_in[6];
  const float* bk = (const float*)d_in[7];
  const float* Wv = (const float*)d_in[8];
  const float* bv = (const float*)d_in[9];
  const float* Wo = (const float*)d_in[10];
  const float* bo = (const float*)d_in[11];
  float* out = (float*)d_out;

  const size_t BSD = (size_t)BATCH * SEQ * D_MODEL;   // 4,194,304
  const size_t DD  = (size_t)D_MODEL * D_MODEL;       // 1,048,576

  bf16_t* qbf   = (bf16_t*)d_ws;          // raw inputs in bf16
  bf16_t* kbf   = qbf + BSD;
  bf16_t* vbf   = kbf + BSD;
  bf16_t* wqb   = vbf + BSD;              // weights in bf16
  bf16_t* wkb   = wqb + DD;
  bf16_t* wvb   = wkb + DD;
  bf16_t* wob   = wvb + DD;
  bf16_t* Qh    = wob + DD;               // [b][h][s][d]
  bf16_t* Kh    = Qh + BSD;               // [b][h][s][d]
  bf16_t* Vt    = Kh + BSD;               // [b][h][d][s]
  bf16_t* joint = Vt + BSD;               // [b][s][d_model]   (total 64 MB)

  const int CT = 256;
  f32_to_bf16_kernel<<<dim3((unsigned)((BSD + CT - 1) / CT)), dim3(CT), 0, stream>>>(q, qbf, (int)BSD);
  f32_to_bf16_kernel<<<dim3((unsigned)((BSD + CT - 1) / CT)), dim3(CT), 0, stream>>>(k, kbf, (int)BSD);
  f32_to_bf16_kernel<<<dim3((unsigned)((BSD + CT - 1) / CT)), dim3(CT), 0, stream>>>(v, vbf, (int)BSD);
  f32_to_bf16_kernel<<<dim3((unsigned)((DD + CT - 1) / CT)),  dim3(CT), 0, stream>>>(Wq, wqb, (int)DD);
  f32_to_bf16_kernel<<<dim3((unsigned)((DD + CT - 1) / CT)),  dim3(CT), 0, stream>>>(Wk, wkb, (int)DD);
  f32_to_bf16_kernel<<<dim3((unsigned)((DD + CT - 1) / CT)),  dim3(CT), 0, stream>>>(Wv, wvb, (int)DD);
  f32_to_bf16_kernel<<<dim3((unsigned)((DD + CT - 1) / CT)),  dim3(CT), 0, stream>>>(Wo, wob, (int)DD);

  const int Mproj = BATCH * SEQ;                                  // 4096
  const int tiles = (Mproj / 32) * (D_MODEL / 64);                // 2048 waves, 4 waves/block
  dim3 gblk(tiles / 4), gthr(128);
  gemm_bf16_wmma_kernel<<<gblk, gthr, 0, stream>>>(qbf, wqb, bq, Qh, nullptr, Mproj, D_MODEL, D_MODEL, 1);
  gemm_bf16_wmma_kernel<<<gblk, gthr, 0, stream>>>(kbf, wkb, bk, Kh, nullptr, Mproj, D_MODEL, D_MODEL, 1);
  gemm_bf16_wmma_kernel<<<gblk, gthr, 0, stream>>>(vbf, wvb, bv, Vt, nullptr, Mproj, D_MODEL, D_MODEL, 2);

  mha_flash_attn_kernel<<<dim3(BATCH * N_HEADS * (SEQ / 16) / 4), dim3(128), 0, stream>>>(
      Qh, Kh, Vt, msk, joint);

  gemm_bf16_wmma_kernel<<<gblk, gthr, 0, stream>>>(joint, wob, bo, nullptr, out, Mproj, D_MODEL, D_MODEL, 3);
}
